// SingleHiddenTransitionModel_58884001628271
// MI455X (gfx1250) — compile-verified, matrix operand order checked by reading, not run on hardware
//
#include <hip/hip_runtime.h>
#include <hip/hip_bf16.h>

typedef __bf16 bf16;
typedef bf16  v16bf __attribute__((ext_vector_type(16)));
typedef float v8f   __attribute__((ext_vector_type(8)));

#define BATCH   512
#define TSTEPS  32

// ---------------------------------------------------------------------------
// Async global->LDS copy path (gfx1250, ASYNCcnt-tracked DMA). Guarded so the
// code still builds (register-staged pipeline) if the builtin is absent.
// Probe-learned prototype: (v4i __device__* src, v4i __shared__* lds, imm, imm)
// ---------------------------------------------------------------------------
#if defined(__AMDGCN__) && __has_builtin(__builtin_amdgcn_global_load_async_to_lds_b128)
#define USE_ASYNC_LDS 1
#else
#define USE_ASYNC_LDS 0
#endif

#if USE_ASYNC_LDS
typedef int vi4 __attribute__((vector_size(16)));
typedef __attribute__((address_space(1))) vi4 gvi4;   // global
typedef __attribute__((address_space(3))) vi4 lvi4;   // LDS

__device__ __forceinline__ void async_cp16(const unsigned* g, unsigned* l) {
  __builtin_amdgcn_global_load_async_to_lds_b128((gvi4*)g, (lvi4*)l, 0, 0);
}
__device__ __forceinline__ void async_wait0() {
  asm volatile("s_wait_asynccnt 0x0" ::: "memory");
}
#endif

// ---------------------------------------------------------------------------
// Build a 16x32 bf16 WMMA fragment from LDS.
// Layout per CDNA5 ISA 7.12.2 (16-bit A-matrix 16x32):
//   lanes 0-15 : K base 0 ; lanes 16-31 : K base 8
//   VGPR 0..3 hold K = kb+0..kb+7 (pairs), VGPR 4..7 hold K = kb+16..kb+23
// LDS stores each 32-elem K-row as 20 uints (40 bf16, padded, 16B aligned).
// ---------------------------------------------------------------------------
__device__ __forceinline__ v16bf frag_from_lds(const unsigned* lds, int rowU, int kbU) {
  uint4 lo = *reinterpret_cast<const uint4*>(lds + rowU + kbU);
  uint4 hi = *reinterpret_cast<const uint4*>(lds + rowU + kbU + 8);
  union { unsigned u[8]; v16bf v; } u_;
  u_.u[0] = lo.x; u_.u[1] = lo.y; u_.u[2] = lo.z; u_.u[3] = lo.w;
  u_.u[4] = hi.x; u_.u[5] = hi.y; u_.u[6] = hi.z; u_.u[7] = hi.w;
  return u_.v;
}

// ---------------------------------------------------------------------------
// bf16 WMMA GEMM:  C[M,N] = act(A[M,K] @ W + bias)
//   A  : [M][K] bf16 row-major (activations)
//   Bt : [N][K] bf16 (weight, pre-transposed once per call)
// Block tile 128x128, 256 threads = 8 wave32 waves (2 M x 4 N), each wave
// computes 64x32 = 4x2 tiles of v_wmma_f32_16x16x32_bf16.
// Double-buffered LDS; tile kt+1 is filled (async DMA or reg-staged with the
// ds_store sunk below the WMMAs) while tile kt is computed; 1 barrier/iter.
// ACT: 0 = identity, 1 = relu.  Cf (f32) and/or Cb (bf16) outputs optional.
// ---------------------------------------------------------------------------
template <int ACT>
__global__ __launch_bounds__(256)
void gemm_bf16(const bf16* __restrict__ A, const bf16* __restrict__ Bt,
               const float* __restrict__ bias,
               float* __restrict__ Cf, bf16* __restrict__ Cb,
               int M, int N, int K) {
  __shared__ __align__(16) unsigned sA[2][128 * 20];
  __shared__ __align__(16) unsigned sB[2][128 * 20];

  const int tid  = threadIdx.x;
  const int lane = tid & 31;
  const int wave = tid >> 5;
  const int wm   = wave >> 2;          // 0..1 -> 64-row slab
  const int wn   = wave & 3;           // 0..3 -> 32-col slab
  const int mBlk = blockIdx.y * 128;
  const int nBlk = blockIdx.x * 128;
  const int Ku   = K >> 1;             // row stride in uints (both A and Bt)

  const unsigned* Ag = reinterpret_cast<const unsigned*>(A);
  const unsigned* Bg = reinterpret_cast<const unsigned*>(Bt);

  const int ktiles = K >> 5;
  const int srow  = tid >> 1;          // staging: 2 threads per 32-elem row
  const int shalf = tid & 1;

  const unsigned* baseA = Ag + (size_t)(mBlk + srow) * Ku + shalf * 8;
  const unsigned* baseB = Bg + (size_t)(nBlk + srow) * Ku + shalf * 8;
  unsigned* dA0 = &sA[0][srow * 20 + shalf * 8];
  unsigned* dB0 = &sB[0][srow * 20 + shalf * 8];
  const int bufStride = 128 * 20;

  v8f acc[4][2];
#pragma unroll
  for (int i = 0; i < 4; ++i)
#pragma unroll
    for (int j = 0; j < 2; ++j) acc[i][j] = (v8f)0.0f;

  // ---- prologue: fill buffer 0 with tile 0 ----
#if USE_ASYNC_LDS
  async_cp16(baseA, dA0);  async_cp16(baseA + 4, dA0 + 4);
  async_cp16(baseB, dB0);  async_cp16(baseB + 4, dB0 + 4);
  async_wait0();
#else
  *(uint4*)dA0       = *(const uint4*)baseA;
  *(uint4*)(dA0 + 4) = *(const uint4*)(baseA + 4);
  *(uint4*)dB0       = *(const uint4*)baseB;
  *(uint4*)(dB0 + 4) = *(const uint4*)(baseB + 4);
#endif
  __syncthreads();

  const int kbU = (lane < 16) ? 0 : 4;  // K-base in uints per ISA layout
  for (int kt = 0; kt < ktiles; ++kt) {
    const int cur = kt & 1;
    const bool haveNext = (kt + 1 < ktiles);
    const unsigned* srcA = baseA + (kt + 1) * 16;
    const unsigned* srcB = baseB + (kt + 1) * 16;
    unsigned* dA = dA0 + (cur ^ 1) * bufStride;
    unsigned* dB = dB0 + (cur ^ 1) * bufStride;

#if USE_ASYNC_LDS
    if (haveNext) {               // fire-and-forget DMA, overlaps WMMAs below
      async_cp16(srcA, dA);  async_cp16(srcA + 4, dA + 4);
      async_cp16(srcB, dB);  async_cp16(srcB + 4, dB + 4);
      if (kt + 2 < ktiles) {
        __builtin_prefetch(srcA + 16, 0, 0);
        __builtin_prefetch(srcB + 16, 0, 0);
      }
    }
#else
    uint4 ra0 = {}, ra1 = {}, rb0 = {}, rb1 = {};
    if (haveNext) {               // issue loads now; ds_store sunk below WMMAs
      ra0 = *(const uint4*)srcA;
      ra1 = *(const uint4*)(srcA + 4);
      rb0 = *(const uint4*)srcB;
      rb1 = *(const uint4*)(srcB + 4);
      if (kt + 2 < ktiles) {
        __builtin_prefetch(srcA + 16, 0, 0);
        __builtin_prefetch(srcB + 16, 0, 0);
      }
    }
#endif

    // ---- compute tile kt ----
    const unsigned* cA = &sA[cur][0];
    const unsigned* cB = &sB[cur][0];
    v16bf af[4];
#pragma unroll
    for (int i = 0; i < 4; ++i) {
      int row = wm * 64 + i * 16 + (lane & 15);
      af[i] = frag_from_lds(cA, row * 20, kbU);
    }
    v16bf bfr[2];
#pragma unroll
    for (int j = 0; j < 2; ++j) {
      int col = wn * 32 + j * 16 + (lane & 15);
      bfr[j] = frag_from_lds(cB, col * 20, kbU);
    }
#pragma unroll
    for (int i = 0; i < 4; ++i)
#pragma unroll
      for (int j = 0; j < 2; ++j)
        acc[i][j] = __builtin_amdgcn_wmma_f32_16x16x32_bf16(
            false, af[i], false, bfr[j], (short)0, acc[i][j], false, false);

    // ---- finish staging tile kt+1, then sync ----
#if USE_ASYNC_LDS
    if (haveNext) async_wait0();
#else
    if (haveNext) {
      *(uint4*)dA       = ra0;
      *(uint4*)(dA + 4) = ra1;
      *(uint4*)dB       = rb0;
      *(uint4*)(dB + 4) = rb1;
    }
#endif
    __syncthreads();
  }

  // ---- epilogue: bias + activation, f32/bf16 stores ----
  const int rHalf = ((lane >> 4) & 1) * 8;   // C/D layout: lanes16-31 hold M+8
#pragma unroll
  for (int j = 0; j < 2; ++j) {
    int col = nBlk + wn * 32 + j * 16 + (lane & 15);
    float bv = bias ? bias[col] : 0.0f;
#pragma unroll
    for (int i = 0; i < 4; ++i) {
      int rowBase = mBlk + wm * 64 + i * 16 + rHalf;
#pragma unroll
      for (int r = 0; r < 8; ++r) {
        float v = acc[i][j][r] + bv;
        if (ACT == 1) v = fmaxf(v, 0.0f);
        size_t idx = (size_t)(rowBase + r) * N + col;
        if (Cf) Cf[idx] = v;
        if (Cb) Cb[idx] = (bf16)v;
      }
    }
  }
}

// ---------------------------------------------------------------------------
// concat two f32 matrices row-wise into one bf16 matrix (zero pad to wd)
// ---------------------------------------------------------------------------
__global__ void concat2_bf16(const float* __restrict__ a, int wa,
                             const float* __restrict__ b, int wb,
                             bf16* __restrict__ dst, int wd, int total) {
  int idx = blockIdx.x * blockDim.x + threadIdx.x;
  if (idx >= total) return;
  int row = idx / wd;
  int c   = idx - row * wd;
  float v = 0.0f;
  if (c < wa)            v = a[(size_t)row * wa + c];
  else if (c < wa + wb)  v = b[(size_t)row * wb + (c - wa)];
  dst[idx] = (bf16)v;
}

// f32 -> bf16 straight conversion (activations used as GEMM A operands)
__global__ void cvt_bf16(const float* __restrict__ src, bf16* __restrict__ dst,
                         int total) {
  int idx = blockIdx.x * blockDim.x + threadIdx.x;
  if (idx >= total) return;
  dst[idx] = (bf16)src[idx];
}

// f32 [K][N] -> bf16 [N][Kpad] transpose with zero K-padding (weights, once/call)
__global__ void cvt_transpose_bf16(const float* __restrict__ src, bf16* __restrict__ dst,
                                   int K, int N, int Kpad, int total) {
  int idx = blockIdx.x * blockDim.x + threadIdx.x;
  if (idx >= total) return;
  int n = idx / Kpad;
  int k = idx - n * Kpad;
  dst[idx] = (k < K) ? (bf16)src[(size_t)k * N + n] : (bf16)0.0f;
}

// GRU gates: gi/gh are [B, 3*1024] (biases pre-added in GEMM epilogue)
__global__ void gru_gate(const float* __restrict__ gi, const float* __restrict__ gh,
                         const float* __restrict__ h,
                         float* __restrict__ hf, bf16* __restrict__ hb,
                         float* __restrict__ out_belief, int total) {
  int idx = blockIdx.x * blockDim.x + threadIdx.x;
  if (idx >= total) return;
  int row = idx >> 10;            // /1024
  int c   = idx & 1023;
  const float* gir = gi + (size_t)row * 3072;
  const float* ghr = gh + (size_t)row * 3072;
  float r = 1.0f / (1.0f + __expf(-(gir[c] + ghr[c])));
  float z = 1.0f / (1.0f + __expf(-(gir[c + 1024] + ghr[c + 1024])));
  float n = tanhf(gir[c + 2048] + r * ghr[c + 2048]);
  float hv = (1.0f - z) * n + z * h[idx];
  hf[idx] = hv;
  hb[idx] = (bf16)hv;
  out_belief[idx] = hv;
}

// distribution head: h2 = [B, 512] (mean | std_raw); state = mean + std*noise
__global__ void dist_head(const float* __restrict__ h2, const float* __restrict__ noise,
                          float* __restrict__ o_state, float* __restrict__ o_mean,
                          float* __restrict__ o_std, float* __restrict__ carry,
                          int total) {
  int idx = blockIdx.x * blockDim.x + threadIdx.x;
  if (idx >= total) return;
  int row = idx >> 8;             // /256
  int c   = idx & 255;
  float m  = h2[(size_t)row * 512 + c];
  float sr = h2[(size_t)row * 512 + 256 + c];
  float sp = (sr > 20.0f) ? sr : log1pf(__expf(sr));
  float sd = sp + 0.1f;
  float st = m + sd * noise[idx];
  o_mean[idx]  = m;
  o_std[idx]   = sd;
  o_state[idx] = st;
  carry[idx]   = st;
}

// ---------------------------------------------------------------------------
extern "C" void kernel_launch(void* const* d_in, const int* in_sizes, int n_in,
                              void* d_out, int out_size, void* d_ws, size_t ws_size,
                              hipStream_t stream) {
  const float* prev_state  = (const float*)d_in[0];
  const float* prev_belief = (const float*)d_in[1];
  const float* poses       = (const float*)d_in[2];
  const float* obs         = (const float*)d_in[3];
  const float* post_noise  = (const float*)d_in[4];
  const float* prior_noise = (const float*)d_in[5];
  const float* w_es   = (const float*)d_in[6];  const float* b_es   = (const float*)d_in[7];
  const float* w_ih   = (const float*)d_in[8];  const float* w_hh   = (const float*)d_in[9];
  const float* b_ih   = (const float*)d_in[10]; const float* b_hh   = (const float*)d_in[11];
  const float* w_ebpo = (const float*)d_in[12]; const float* b_ebpo = (const float*)d_in[13];
  const float* w_spo  = (const float*)d_in[14]; const float* b_spo  = (const float*)d_in[15];
  const float* w_ebpr = (const float*)d_in[16]; const float* b_ebpr = (const float*)d_in[17];
  const float* w_spr  = (const float*)d_in[18]; const float* b_spr  = (const float*)d_in[19];

  // ---- workspace bump allocator ----
  char* ws = (char*)d_ws;
  size_t off = 0;
  auto alloc = [&](size_t bytes) -> char* {
    char* p = ws + off;
    off = (off + bytes + 255) & ~(size_t)255;
    return p;
  };

  // bf16 weight mirrors, stored transposed [N][Kpad]
  bf16* wb_es   = (bf16*)alloc((size_t)1024 * 512 * 2);
  bf16* wb_ih   = (bf16*)alloc((size_t)3072 * 1024 * 2);
  bf16* wb_hh   = (bf16*)alloc((size_t)3072 * 1024 * 2);
  bf16* wb_ebpo = (bf16*)alloc((size_t)1024 * 2048 * 2);
  bf16* wb_spo  = (bf16*)alloc((size_t)512 * 1024 * 2);
  bf16* wb_ebpr = (bf16*)alloc((size_t)1024 * 1056 * 2);  // K padded 1030 -> 1056
  bf16* wb_spr  = (bf16*)alloc((size_t)512 * 1024 * 2);

  // activations / carries
  bf16*  x0_bf     = (bf16*)alloc((size_t)BATCH * 512 * 2);
  bf16*  hidden_bf = (bf16*)alloc((size_t)BATCH * 1024 * 2);
  float* gi_f      = (float*)alloc((size_t)BATCH * 3072 * 4);
  float* gh_f      = (float*)alloc((size_t)BATCH * 3072 * 4);
  float* belief_f  = (float*)alloc((size_t)BATCH * 1024 * 4);
  bf16*  belief_bf = (bf16*)alloc((size_t)BATCH * 1024 * 2);
  bf16*  xpo_bf    = (bf16*)alloc((size_t)BATCH * 2048 * 2);
  bf16*  hp_bf     = (bf16*)alloc((size_t)BATCH * 1024 * 2);
  float* h2po_f    = (float*)alloc((size_t)BATCH * 512 * 4);
  bf16*  xpr_bf    = (bf16*)alloc((size_t)BATCH * 1056 * 2);
  bf16*  hq_bf     = (bf16*)alloc((size_t)BATCH * 1024 * 2);
  float* h2pr_f    = (float*)alloc((size_t)BATCH * 512 * 4);
  float* post_c    = (float*)alloc((size_t)BATCH * 256 * 4);
  float* prior_c   = (float*)alloc((size_t)BATCH * 256 * 4);

  auto cvtT = [&](const float* s, bf16* d, int K, int N, int Kpad) {
    int total = N * Kpad;
    cvt_transpose_bf16<<<(total + 255) / 256, 256, 0, stream>>>(s, d, K, N, Kpad, total);
  };
  auto gemm = [&](const bf16* A, const bf16* Bt, const float* bias,
                  float* Cf, bf16* Cb, int M, int N, int K, int act) {
    dim3 g(N / 128, M / 128), b(256);
    if (act) gemm_bf16<1><<<g, b, 0, stream>>>(A, Bt, bias, Cf, Cb, M, N, K);
    else     gemm_bf16<0><<<g, b, 0, stream>>>(A, Bt, bias, Cf, Cb, M, N, K);
  };

  // ---- one-time per call: transposed bf16 weights + carry init ----
  cvtT(w_es,   wb_es,   512,  1024, 512);
  cvtT(w_ih,   wb_ih,   1024, 3072, 1024);
  cvtT(w_hh,   wb_hh,   1024, 3072, 1024);
  cvtT(w_ebpo, wb_ebpo, 2048, 1024, 2048);
  cvtT(w_spo,  wb_spo,  1024, 512,  1024);
  cvtT(w_ebpr, wb_ebpr, 1030, 1024, 1056);   // zero-pad K 1030..1055
  cvtT(w_spr,  wb_spr,  1024, 512,  1024);

  (void)hipMemcpyAsync(post_c,   prev_state,  (size_t)BATCH * 256 * 4,  hipMemcpyDeviceToDevice, stream);
  (void)hipMemcpyAsync(prior_c,  prev_state,  (size_t)BATCH * 256 * 4,  hipMemcpyDeviceToDevice, stream);
  (void)hipMemcpyAsync(belief_f, prev_belief, (size_t)BATCH * 1024 * 4, hipMemcpyDeviceToDevice, stream);
  {
    int total = BATCH * 1024;
    cvt_bf16<<<(total + 255) / 256, 256, 0, stream>>>(prev_belief, belief_bf, total);
  }

  // ---- output slabs (tuple order of the reference) ----
  float* out = (float*)d_out;
  const size_t SB = (size_t)BATCH * 1024;
  const size_t SS = (size_t)BATCH * 256;
  float* oBel  = out;
  float* oPrS  = oBel  + (size_t)TSTEPS * SB;
  float* oPrM  = oPrS  + (size_t)TSTEPS * SS;
  float* oPrSd = oPrM  + (size_t)TSTEPS * SS;
  float* oPoS  = oPrSd + (size_t)TSTEPS * SS;
  float* oPoM  = oPoS  + (size_t)TSTEPS * SS;
  float* oPoSd = oPoM  + (size_t)TSTEPS * SS;

  // ---- sequential scan over T ----
  for (int t = 0; t < TSTEPS; ++t) {
    const float* obs_t  = obs         + (size_t)t * BATCH * 1024;
    const float* pose_t = poses       + (size_t)t * BATCH * 6;
    const float* pn_t   = post_noise  + (size_t)t * BATCH * 256;
    const float* prn_t  = prior_noise + (size_t)t * BATCH * 256;

    // hidden = relu([post, prior] @ w_es + b_es)
    {
      int total = BATCH * 512;
      concat2_bf16<<<(total + 255) / 256, 256, 0, stream>>>(post_c, 256, prior_c, 256,
                                                            x0_bf, 512, total);
    }
    gemm(x0_bf, wb_es, b_es, nullptr, hidden_bf, BATCH, 1024, 512, 1);

    // GRU: gi = hidden @ w_ih + b_ih ; gh = belief @ w_hh + b_hh
    gemm(hidden_bf, wb_ih, b_ih, gi_f, nullptr, BATCH, 3072, 1024, 0);
    gemm(belief_bf, wb_hh, b_hh, gh_f, nullptr, BATCH, 3072, 1024, 0);
    {
      int total = BATCH * 1024;
      gru_gate<<<(total + 255) / 256, 256, 0, stream>>>(gi_f, gh_f, belief_f,
                                                        belief_f, belief_bf,
                                                        oBel + (size_t)t * SB, total);
    }

    // posterior branch
    {
      int total = BATCH * 2048;
      concat2_bf16<<<(total + 255) / 256, 256, 0, stream>>>(belief_f, 1024, obs_t, 1024,
                                                            xpo_bf, 2048, total);
    }
    gemm(xpo_bf, wb_ebpo, b_ebpo, nullptr, hp_bf, BATCH, 1024, 2048, 1);
    gemm(hp_bf, wb_spo, b_spo, h2po_f, nullptr, BATCH, 512, 1024, 0);
    {
      int total = BATCH * 256;
      dist_head<<<(total + 255) / 256, 256, 0, stream>>>(h2po_f, pn_t,
                                                         oPoS + (size_t)t * SS,
                                                         oPoM + (size_t)t * SS,
                                                         oPoSd + (size_t)t * SS,
                                                         post_c, total);
    }

    // prior branch (pose concat, zero-padded K 1030 -> 1056)
    {
      int total = BATCH * 1056;
      concat2_bf16<<<(total + 255) / 256, 256, 0, stream>>>(belief_f, 1024, pose_t, 6,
                                                            xpr_bf, 1056, total);
    }
    gemm(xpr_bf, wb_ebpr, b_ebpr, nullptr, hq_bf, BATCH, 1024, 1056, 1);
    gemm(hq_bf, wb_spr, b_spr, h2pr_f, nullptr, BATCH, 512, 1024, 0);
    {
      int total = BATCH * 256;
      dist_head<<<(total + 255) / 256, 256, 0, stream>>>(h2pr_f, prn_t,
                                                         oPrS + (size_t)t * SS,
                                                         oPrM + (size_t)t * SS,
                                                         oPrSd + (size_t)t * SS,
                                                         prior_c, total);
    }
  }
}
